// SwinBlock_71356586655756
// MI455X (gfx1250) — compile-verified, matrix operand order checked by reading, not run on hardware
//
#include <hip/hip_runtime.h>
#include <hip/hip_bf16.h>
#include <math.h>

// ---------------------------------------------------------------------------
// Swin block for MI455X (gfx1250, wave32, WMMA + Tensor Data Mover).
// GEMMs: v_wmma_f32_16x16x32_bf16 (bf16 A/B, f32 accumulate).
// WMMA operands come from LDS as 2x ds_load_b128 per lane; A operands are
// row-major [M][K], B operands stored transposed [N][K] so one loader serves
// both.  Weight tiles are staged into LDS by TDM (tensor_load_to_lds) from
// pre-converted bf16 [N][K] copies in the workspace; the TDM pad feature
// inserts the 16B row padding to produce the ld=40 LDS layout directly.
// ---------------------------------------------------------------------------

typedef __bf16 bf16;
typedef __attribute__((ext_vector_type(16))) __bf16 v16bf;
typedef __attribute__((ext_vector_type(8)))  __bf16 v8bf;
typedef __attribute__((ext_vector_type(8)))  float  v8f;
typedef __attribute__((ext_vector_type(4)))  unsigned int u32x4;
typedef __attribute__((ext_vector_type(8)))  int i32x8;
typedef __attribute__((ext_vector_type(4)))  int i32x4;

__device__ __forceinline__ v8f zero8() {
  v8f z;
#pragma unroll
  for (int i = 0; i < 8; ++i) z[i] = 0.f;
  return z;
}

// Fragment loader (A from [M][K] storage, B from [N][K] transposed storage).
// lane l -> row = base + (l&15); element j -> k = kb + (j&7) + 8*(l>>4) + (j>>3)*16
// Requires ld % 8 == 0 and kb % 8 == 0 so both halves are 16-byte aligned.
__device__ __forceinline__ v16bf frag_ld(const bf16* p, int ld, int row, int kb) {
  const int lane = threadIdx.x & 31;
  const bf16* q = p + (row + (lane & 15)) * ld + kb + 8 * (lane >> 4);
  v8bf lo = *(const v8bf*)q;          // k = kb+8*half .. +7
  v8bf hi = *(const v8bf*)(q + 16);   // k = kb+16+8*half .. +7
  return __builtin_shufflevector(lo, hi, 0, 1, 2, 3, 4, 5, 6, 7,
                                 8, 9, 10, 11, 12, 13, 14, 15);
}

__device__ __forceinline__ v8f wmma_bf16(v16bf a, v16bf b, v8f c) {
  return __builtin_amdgcn_wmma_f32_16x16x32_bf16(false, a, false, b, (short)0, c,
                                                 false, false);
}

// Byte offset of a __shared__ object inside the LDS segment (generic -> AS3).
__device__ __forceinline__ unsigned lds_offset(const void* p) {
  return (unsigned)(unsigned long long)(const __attribute__((address_space(3))) char*)p;
}

// TDM: load a 2D bf16 tile [tile_n rows x 32 k-elems] from a [.,stride_k]
// row-major bf16 tensor into LDS with 16B padding per 64B row (-> ld = 40).
// Issue from ONE wave only; completion via s_wait_tensorcnt + barrier.
__device__ __forceinline__ void tdm_load_tile(unsigned lds_off, const bf16* gptr,
                                              int tile_n, int stride_k) {
  unsigned long long ga = (unsigned long long)gptr;
  u32x4 g0;
  g0[0] = 1u;                                            // count=1, user D#
  g0[1] = lds_off;                                       // LDS byte address
  g0[2] = (unsigned)(ga & 0xFFFFFFFFu);                  // global addr [31:0]
  g0[3] = (unsigned)((ga >> 32) & 0x01FFFFFFu)           // global addr [56:32]
          | 0x80000000u;                                 // type = 2 ("image")
  i32x8 g1;
  g1[0] = (1 << 16)        // data_size = 1 -> 2 bytes
        | (1 << 20)        // pad_enable
        | (3 << 22)        // pad_interval code 3 = 16 DWORDs (64B)
        | (3 << 25);       // pad_amount  code 3 = 4 DWORDs (16B)
  g1[1] = (32 << 16);                      // tensor_dim0 = 32 (bits 79:48)
  g1[2] = (tile_n & 0xFFFF) << 16;         // tensor_dim1 lo (bits 111:80)
  g1[3] = ((tile_n >> 16) & 0xFFFF) | (32 << 16);  // dim1 hi | tile_dim0 = 32
  g1[4] = tile_n & 0xFFFF;                 // tile_dim1 (tile_dim2 = 0)
  g1[5] = stride_k;                        // tensor_dim0_stride lo32
  g1[6] = 0;                               // stride hi | dim1_stride lo
  g1[7] = 0;
  i32x4 gz;
  gz[0] = 0; gz[1] = 0; gz[2] = 0; gz[3] = 0;
  i32x8 gz8;
#pragma unroll
  for (int i = 0; i < 8; ++i) gz8[i] = 0;
  __builtin_amdgcn_tensor_load_to_lds(g0, g1, gz, gz, gz8, 0);
}

// ---------------------------------------------------------------------------
// Kernel 0: one-time fp32 -> bf16 transposed weight conversion into workspace.
//   wT layout (elements): [0)      qkvT [384][128]
//                         [49152)  projT [128][128]
//                         [65536)  w1T  [512][128]
//                         [131072) w2T  [128][512]   (end 196608)
// ---------------------------------------------------------------------------
__global__ void __launch_bounds__(256)
convert_weights(const float* __restrict__ qkv_w, const float* __restrict__ proj_w,
                const float* __restrict__ w1, const float* __restrict__ w2,
                bf16* __restrict__ wT) {
  int i = blockIdx.x * 256 + threadIdx.x;
  if (i < 49152) {
    int n = i >> 7, k = i & 127;
    wT[i] = (bf16)qkv_w[k * 384 + n];
  } else if (i < 65536) {
    int j = i - 49152; int n = j >> 7, k = j & 127;
    wT[i] = (bf16)proj_w[k * 128 + n];
  } else if (i < 131072) {
    int j = i - 65536; int n = j >> 7, k = j & 127;
    wT[i] = (bf16)w1[k * 512 + n];
  } else if (i < 196608) {
    int j = i - 131072; int n = j >> 9, k = j & 511;
    wT[i] = (bf16)w2[k * 128 + n];
  }
}

// ---------------------------------------------------------------------------
// Kernel 1: LN1 + windowed attention + proj + residual.  One window / block.
//   grid = 4096 windows (B=4 x 32 x 32), block = 256 threads (8 waves).
//   Writes xmid[token][c] (token-major fp32) to workspace.
// ---------------------------------------------------------------------------
__global__ void __launch_bounds__(256)
swin_attn_kernel(const float* __restrict__ x, const bf16* __restrict__ wqkvT,
                 const float* __restrict__ qkv_b, const bf16* __restrict__ wprojT,
                 const float* __restrict__ proj_b, const float* __restrict__ rel_bias,
                 const float* __restrict__ ln1_g, const float* __restrict__ ln1_b,
                 float* __restrict__ xmid) {
  __shared__ __align__(16) bf16 sA[64 * 136];     // LN1 out, rows 49..63 zero
  __shared__ __align__(16) bf16 sWt[384 * 40];    // TDM-staged weights [N][K]
  __shared__ __align__(16) bf16 sQ[64 * 136];     // Q (pre-scaled), [tok][dim]
  __shared__ __align__(16) bf16 sK[64 * 136];     // K, [tok][dim]  (= B^T of QK^T)
  __shared__ __align__(16) bf16 sVT[128 * 72];    // V^T, [dim][tok] (= B^T of PV)
  __shared__ __align__(16) bf16 sP[4 * 64 * 72];  // softmax probs per head
  __shared__ __align__(16) bf16 sO[64 * 136];     // attention output (heads concat)

  const int tid  = threadIdx.x;
  const int lane = tid & 31;
  const int wv   = tid >> 5;
  const int half = lane >> 4;
  const int jcol = lane & 15;
  const unsigned sWt_off = lds_offset(sWt);

  const int win = blockIdx.x;
  const int b   = win >> 10;          // / (32*32)
  const int wh  = (win >> 5) & 31;
  const int ww  = win & 31;

  // -------- Phase 1: LN1 on 49 tokens (one wave per token) ----------------
  for (int it = 0; it < 8; ++it) {
    int m = wv + 8 * it;              // 0..63
    if (m < 49) {
      int ty = m / 7, tx = m - 7 * ty;
      int gh = wh * 7 + ty, gw = ww * 7 + tx;
      float v[4];
      float s = 0.f, ss = 0.f;
#pragma unroll
      for (int j = 0; j < 4; ++j) {
        int c = lane + 32 * j;
        float t = x[(((size_t)b * 128 + c) * 224 + gh) * 224 + gw];
        v[j] = t; s += t; ss += t * t;
      }
#pragma unroll
      for (int o = 16; o > 0; o >>= 1) {
        s  += __shfl_xor(s, o, 32);
        ss += __shfl_xor(ss, o, 32);
      }
      float mean = s * (1.f / 128.f);
      float var  = ss * (1.f / 128.f) - mean * mean;
      float rstd = rsqrtf(var + 1e-5f);
#pragma unroll
      for (int j = 0; j < 4; ++j) {
        int c = lane + 32 * j;
        sA[m * 136 + c] = (bf16)((v[j] - mean) * rstd * ln1_g[c] + ln1_b[c]);
      }
    } else {
#pragma unroll
      for (int j = 0; j < 4; ++j) sA[m * 136 + lane + 32 * j] = (bf16)0.f;
    }
  }

  // -------- Phase 2: QKV = A(64x128) @ qkv_w(128x384) ---------------------
  const int mt    = wv & 3;           // M tile (rows mt*16..)
  const int nhalf = wv >> 2;          // which 192-col half of the 384 outputs
  v8f acc[12];
#pragma unroll
  for (int i = 0; i < 12; ++i) acc[i] = zero8();

  for (int kc = 0; kc < 4; ++kc) {
    __syncthreads();                  // previous chunk fully consumed
    if (wv == 0) {
      tdm_load_tile(sWt_off, wqkvT + kc * 32, 384, 128);
      __builtin_amdgcn_s_wait_tensorcnt(0);
    }
    __syncthreads();                  // chunk visible to all waves
    v16bf af = frag_ld(sA, 136, mt * 16, kc * 32);
#pragma unroll
    for (int nt = 0; nt < 12; ++nt) {
      v16bf bw = frag_ld(sWt, 40, nhalf * 192 + nt * 16, 0);
      acc[nt] = wmma_bf16(af, bw, acc[nt]);
    }
  }
  __syncthreads();

  // Route Q (scaled), K, V^T into LDS.
  const float qscale = 0.17677669529663687f;   // 1/sqrt(32)
#pragma unroll
  for (int nt = 0; nt < 12; ++nt) {
    int n = nhalf * 192 + nt * 16 + jcol;
    float bias = qkv_b[n];
    if (n < 256) {                     // Q and K: row-major [tok][dim]
#pragma unroll
      for (int r = 0; r < 8; ++r) {
        int m = mt * 16 + r + 8 * half;
        float val = acc[nt][r] + bias;
        if (n < 128) sQ[m * 136 + n]         = (bf16)(val * qscale);
        else         sK[m * 136 + (n - 128)] = (bf16)val;
      }
    } else {                           // V^T: [dim][tok]; 8 contiguous tokens
      v8bf pk;
#pragma unroll
      for (int r = 0; r < 8; ++r) pk[r] = (bf16)(acc[nt][r] + bias);
      *(v8bf*)&sVT[(n - 256) * 72 + mt * 16 + 8 * half] = pk;
    }
  }
  __syncthreads();

  // -------- Phase 3: S = Q K^T (+bias), softmax -> P ----------------------
  const int hd_ = wv >> 1;            // head 0..3
  const int mh  = wv & 1;             // pair of M tiles
  v8f sc[2][4];
#pragma unroll
  for (int i = 0; i < 2; ++i)
#pragma unroll
    for (int j = 0; j < 4; ++j) sc[i][j] = zero8();

#pragma unroll
  for (int i = 0; i < 2; ++i) {
    v16bf aq = frag_ld(sQ, 136, (mh * 2 + i) * 16, hd_ * 32);
#pragma unroll
    for (int jt = 0; jt < 4; ++jt) {
      v16bf bk = frag_ld(sK, 136, jt * 16, hd_ * 32);   // B^T = K row-major
      sc[i][jt] = wmma_bf16(aq, bk, sc[i][jt]);
    }
  }

#pragma unroll
  for (int i = 0; i < 2; ++i) {
#pragma unroll
    for (int r = 0; r < 8; ++r) {
      int m = (mh * 2 + i) * 16 + r + 8 * half;
      float sv[4];
#pragma unroll
      for (int jt = 0; jt < 4; ++jt) {
        int j = jt * 16 + jcol;
        float t = sc[i][jt][r];
        if (j < 49 && m < 49) {
          int yi = m / 7, xi = m - 7 * yi;
          int yj = j / 7, xj = j - 7 * yj;
          t += rel_bias[hd_ * 169 + (yi - yj + 6) * 13 + (xi - xj + 6)];
        } else {
          t = -1e30f;                  // mask padded cols / rows
        }
        sv[jt] = t;
      }
      float mx = fmaxf(fmaxf(sv[0], sv[1]), fmaxf(sv[2], sv[3]));
#pragma unroll
      for (int o = 1; o < 16; o <<= 1) mx = fmaxf(mx, __shfl_xor(mx, o, 32));
      float pv[4];
      float sum = 0.f;
#pragma unroll
      for (int jt = 0; jt < 4; ++jt) {
        float e = (sv[jt] > -1e29f) ? __expf(sv[jt] - mx) : 0.f;
        pv[jt] = e; sum += e;
      }
#pragma unroll
      for (int o = 1; o < 16; o <<= 1) sum += __shfl_xor(sum, o, 32);
      float inv = (m < 49) ? 1.f / sum : 0.f;
#pragma unroll
      for (int jt = 0; jt < 4; ++jt)
        sP[(hd_ * 64 + m) * 72 + jt * 16 + jcol] = (bf16)(pv[jt] * inv);
    }
  }
  __syncthreads();

  // -------- Phase 4: O = P V per head -------------------------------------
  v8f oc[2][2];
#pragma unroll
  for (int i = 0; i < 2; ++i)
#pragma unroll
    for (int j = 0; j < 2; ++j) oc[i][j] = zero8();

#pragma unroll
  for (int kc = 0; kc < 2; ++kc) {
#pragma unroll
    for (int i = 0; i < 2; ++i) {
      v16bf ap = frag_ld(sP + hd_ * 64 * 72, 72, (mh * 2 + i) * 16, kc * 32);
#pragma unroll
      for (int nt = 0; nt < 2; ++nt) {
        v16bf bv = frag_ld(sVT, 72, hd_ * 32 + nt * 16, kc * 32); // B^T = V^T
        oc[i][nt] = wmma_bf16(ap, bv, oc[i][nt]);
      }
    }
  }
#pragma unroll
  for (int i = 0; i < 2; ++i)
#pragma unroll
    for (int nt = 0; nt < 2; ++nt) {
      int n = hd_ * 32 + nt * 16 + jcol;
#pragma unroll
      for (int r = 0; r < 8; ++r) {
        int m = (mh * 2 + i) * 16 + r + 8 * half;
        sO[m * 136 + n] = (bf16)oc[i][nt][r];
      }
    }
  __syncthreads();

  // -------- Phase 5: proj + bias + shortcut -> xmid (token-major) ---------
  const int nh2 = wv >> 2;
  v8f pr[4];
#pragma unroll
  for (int i = 0; i < 4; ++i) pr[i] = zero8();

  for (int kc = 0; kc < 4; ++kc) {
    __syncthreads();
    if (wv == 0) {
      tdm_load_tile(sWt_off, wprojT + kc * 32, 128, 128);
      __builtin_amdgcn_s_wait_tensorcnt(0);
    }
    __syncthreads();
    v16bf ao = frag_ld(sO, 136, mt * 16, kc * 32);
#pragma unroll
    for (int nt = 0; nt < 4; ++nt) {
      v16bf bw = frag_ld(sWt, 40, nh2 * 64 + nt * 16, 0);
      pr[nt] = wmma_bf16(ao, bw, pr[nt]);
    }
  }

#pragma unroll
  for (int nt = 0; nt < 4; ++nt) {
    int c = nh2 * 64 + nt * 16 + jcol;
    float pb = proj_b[c];
#pragma unroll
    for (int r = 0; r < 8; ++r) {
      int m = mt * 16 + r + 8 * half;
      if (m < 49) {
        int ty = m / 7, tx = m - 7 * ty;
        int gh = wh * 7 + ty, gw = ww * 7 + tx;
        size_t tok = ((size_t)b * 224 + gh) * 224 + gw;
        float sc0 = x[(((size_t)b * 128 + c) * 224 + gh) * 224 + gw];
        xmid[tok * 128 + c] = pr[nt][r] + pb + sc0;
      }
    }
  }
}

// ---------------------------------------------------------------------------
// Kernel 2: LN2 + MLP (128->512 GELU ->128) + residual, scatter to (B,C,H,W).
//   grid = 3136 blocks of 64 tokens, block = 256 threads (8 waves).
// ---------------------------------------------------------------------------
__global__ void __launch_bounds__(256)
swin_mlp_kernel(const float* __restrict__ xmid, const float* __restrict__ ln2_g,
                const float* __restrict__ ln2_b, const bf16* __restrict__ w1T,
                const float* __restrict__ b1, const bf16* __restrict__ w2T,
                const float* __restrict__ b2, float* __restrict__ out) {
  __shared__ __align__(16) bf16 sA[64 * 136];   // LN2 output
  __shared__ __align__(16) bf16 sWt[512 * 40];  // TDM-staged weights [N][K]
  __shared__ __align__(16) bf16 sH[64 * 520];   // hidden after GELU

  const int tid  = threadIdx.x;
  const int lane = tid & 31;
  const int wv   = tid >> 5;
  const int half = lane >> 4;
  const int jcol = lane & 15;
  const long t0  = (long)blockIdx.x * 64;
  const unsigned sWt_off = lds_offset(sWt);

  // -------- LN2 (one wave per token) --------------------------------------
  for (int it = 0; it < 8; ++it) {
    int m = wv + 8 * it;
    float v[4];
    float s = 0.f, ss = 0.f;
#pragma unroll
    for (int j = 0; j < 4; ++j) {
      int c = lane + 32 * j;
      float t = xmid[(t0 + m) * 128 + c];
      v[j] = t; s += t; ss += t * t;
    }
#pragma unroll
    for (int o = 16; o > 0; o >>= 1) {
      s  += __shfl_xor(s, o, 32);
      ss += __shfl_xor(ss, o, 32);
    }
    float mean = s * (1.f / 128.f);
    float var  = ss * (1.f / 128.f) - mean * mean;
    float rstd = rsqrtf(var + 1e-5f);
#pragma unroll
    for (int j = 0; j < 4; ++j) {
      int c = lane + 32 * j;
      sA[m * 136 + c] = (bf16)((v[j] - mean) * rstd * ln2_g[c] + ln2_b[c]);
    }
  }

  // -------- GEMM1: H = A(64x128) @ w1(128x512), bias + exact GELU ---------
  const int mt    = wv & 3;
  const int nhalf = wv >> 2;
  v8f h[16];
#pragma unroll
  for (int i = 0; i < 16; ++i) h[i] = zero8();

  for (int kc = 0; kc < 4; ++kc) {
    __syncthreads();
    if (wv == 0) {
      tdm_load_tile(sWt_off, w1T + kc * 32, 512, 128);
      __builtin_amdgcn_s_wait_tensorcnt(0);
    }
    __syncthreads();
    v16bf af = frag_ld(sA, 136, mt * 16, kc * 32);
#pragma unroll
    for (int nt = 0; nt < 16; ++nt) {
      v16bf bw = frag_ld(sWt, 40, nhalf * 256 + nt * 16, 0);
      h[nt] = wmma_bf16(af, bw, h[nt]);
    }
  }
  __syncthreads();

#pragma unroll
  for (int nt = 0; nt < 16; ++nt) {
    int n = nhalf * 256 + nt * 16 + jcol;
    float bb = b1[n];
#pragma unroll
    for (int r = 0; r < 8; ++r) {
      int m = mt * 16 + r + 8 * half;
      float v2 = h[nt][r] + bb;
      float g  = 0.5f * v2 * (1.f + erff(v2 * 0.70710678118f));  // exact GELU
      sH[m * 520 + n] = (bf16)g;
    }
  }
  __syncthreads();

  // -------- GEMM2: Y = H(64x512) @ w2(512x128) + b2 + residual ------------
  const int nh2 = wv >> 2;
  v8f y[4];
#pragma unroll
  for (int i = 0; i < 4; ++i) y[i] = zero8();

  for (int kc = 0; kc < 16; ++kc) {
    __syncthreads();
    if (wv == 0) {
      tdm_load_tile(sWt_off, w2T + kc * 32, 128, 512);
      __builtin_amdgcn_s_wait_tensorcnt(0);
    }
    __syncthreads();
    v16bf af = frag_ld(sH, 520, mt * 16, kc * 32);
#pragma unroll
    for (int nt = 0; nt < 4; ++nt) {
      v16bf bw = frag_ld(sWt, 40, nh2 * 64 + nt * 16, 0);
      y[nt] = wmma_bf16(af, bw, y[nt]);
    }
  }

#pragma unroll
  for (int nt = 0; nt < 4; ++nt) {
    int c = nh2 * 64 + nt * 16 + jcol;
    float bb = b2[c];
#pragma unroll
    for (int r = 0; r < 8; ++r) {
      int m = mt * 16 + r + 8 * half;
      long T = t0 + m;
      float val = y[nt][r] + bb + xmid[T * 128 + c];
      long bz  = T / (224 * 224);
      long rem = T - bz * (224 * 224);
      long gh  = rem / 224, gw = rem - gh * 224;
      out[((bz * 128 + c) * 224 + gh) * 224 + gw] = val;
    }
  }
}

// ---------------------------------------------------------------------------
extern "C" void kernel_launch(void* const* d_in, const int* in_sizes, int n_in,
                              void* d_out, int out_size, void* d_ws, size_t ws_size,
                              hipStream_t stream) {
  (void)in_sizes; (void)n_in; (void)out_size; (void)ws_size;
  const float* x      = (const float*)d_in[0];
  const float* qkv_w  = (const float*)d_in[1];
  const float* qkv_b  = (const float*)d_in[2];
  const float* proj_w = (const float*)d_in[3];
  const float* proj_b = (const float*)d_in[4];
  const float* rel_b  = (const float*)d_in[5];
  const float* ln1_g  = (const float*)d_in[6];
  const float* ln1_bb = (const float*)d_in[7];
  const float* ln2_g  = (const float*)d_in[8];
  const float* ln2_bb = (const float*)d_in[9];
  const float* w1     = (const float*)d_in[10];
  const float* b1     = (const float*)d_in[11];
  const float* w2     = (const float*)d_in[12];
  const float* b2     = (const float*)d_in[13];
  float* out  = (float*)d_out;

  float* xmid = (float*)d_ws;                       // 200704*128 fp32 = 98 MiB
  bf16*  wT   = (bf16*)((char*)d_ws + (size_t)200704 * 128 * 4);
  bf16*  wqkvT  = wT;                               // [384][128]
  bf16*  wprojT = wT + 49152;                       // [128][128]
  bf16*  w1T    = wT + 65536;                       // [512][128]
  bf16*  w2T    = wT + 131072;                      // [128][512]

  convert_weights<<<768, 256, 0, stream>>>(qkv_w, proj_w, w1, w2, wT);
  swin_attn_kernel<<<4096, 256, 0, stream>>>(x, wqkvT, qkv_b, wprojT, proj_b,
                                             rel_b, ln1_g, ln1_bb, xmid);
  swin_mlp_kernel<<<3136, 256, 0, stream>>>(xmid, ln2_g, ln2_bb, w1T, b1, w2T,
                                            b2, out);
}
